// GATLayer_71201967833910
// MI455X (gfx1250) — matrix operand order repaired and checked by reading, not compiled, not run on hardware
//
#include <hip/hip_runtime.h>
#include <hip/hip_bf16.h>

// ---------------------------------------------------------------------------
// GAT layer, MI455X (gfx1250, wave32, WMMA + async global->LDS).
// N=4096 nodes, F=128, H=8 heads, D=8. Dense binary adjacency mask.
//
// out[i,h,:] = relu( (sum_j A[i,j]*e[h,i,j]*feats[j,h,:]) / (sum_j A[i,j]*e[h,i,j]) + bias )
//   e[h,i,j] = exp(leaky_relu(a_s[i,h] + a_n[j,h], 0.2))
// (exp(-1e10) == 0 in fp32, so the masked-softmax collapses to this form.)
//
// Aggregation GEMM: v_wmma_f32_16x16x32_bf16. The 16x32 M-tile (mask*exp) is
// built in VALU each iteration (a_s/a_n pre-scaled by log2e so the kernel is
// add + max + v_exp_f32 per element). A-tile and a_n-chunk are staged in LDS
// with GLOBAL_LOAD_ASYNC_TO_LDS (ASYNCcnt) so the staging latency overlaps
// the exp/WMMA work; column 8 of B is all-ones so the softmax denominator Z
// accumulates in output column 8 of the same WMMA chain.
// ---------------------------------------------------------------------------

typedef __attribute__((ext_vector_type(16))) __bf16          v16bf;
typedef __attribute__((ext_vector_type(8)))  float           v8f;
typedef __attribute__((ext_vector_type(16))) unsigned short  v16u;

#define NNODES 4096
#define NF     128
#define NH     8
#define ND     8
#define LOG2E  1.44269504088896340736f

static __device__ __forceinline__ unsigned short f2bf(float f) {
    // round-to-nearest-even float -> bf16 bits
    unsigned int u = __float_as_uint(f);
    u += 0x7FFFu + ((u >> 16) & 1u);
    return (unsigned short)(u >> 16);
}

// mask * exp2( max(y, 0.2y) )  with y already scaled by log2(e)
static __device__ __forceinline__ __bf16 melem(float mask, float y) {
    float l = fmaxf(y, 0.2f * y);                 // leaky_relu(0.2), one v_max
    return (__bf16)(mask * __builtin_amdgcn_exp2f(l));
}

// async global -> LDS copies (CDNA5; ASYNCcnt-tracked, no VGPR round trip)
static __device__ __forceinline__ void async_lds_b64(unsigned lds_off, const void* gaddr) {
    asm volatile("global_load_async_to_lds_b64 %0, %1, off"
                 :: "v"(lds_off), "v"(gaddr) : "memory");
}
static __device__ __forceinline__ void async_lds_b32(unsigned lds_off, const void* gaddr) {
    asm volatile("global_load_async_to_lds_b32 %0, %1, off"
                 :: "v"(lds_off), "v"(gaddr) : "memory");
}
static __device__ __forceinline__ void wait_async0() {
    asm volatile("s_wait_asynccnt 0x0" ::: "memory");
}

// ---------------------------------------------------------------------------
// Kernel 1: feats = X@W (fp32), emit transposed bf16 featsT[h][16][N]
// (col 8 = 1.0 for the Z trick, cols 9..15 = 0), and log2e-scaled
// a_sT/a_nT[h][N]. One block per node, 64 threads.
// ---------------------------------------------------------------------------
__global__ __launch_bounds__(64)
void gat_feats_kernel(const float* __restrict__ X,
                      const float* __restrict__ W,
                      const float* __restrict__ att_self,
                      const float* __restrict__ att_neigh,
                      unsigned short* __restrict__ featsT,
                      float* __restrict__ a_sT,
                      float* __restrict__ a_nT) {
    const int i = blockIdx.x;      // node
    const int t = threadIdx.x;     // 0..63 -> output column
    __shared__ float xrow[NF];
    __shared__ float frow[NH * ND];

    xrow[t]      = X[(size_t)i * NF + t];
    xrow[t + 64] = X[(size_t)i * NF + t + 64];
    __syncthreads();

    float acc = 0.0f;
#pragma unroll 8
    for (int k = 0; k < NF; ++k)
        acc = fmaf(xrow[k], W[(size_t)k * (NH * ND) + t], acc);
    frow[t] = acc;

    const int h = t >> 3;
    const int d = t & 7;
    // feature columns 0..7 of head h (transposed along nodes)
    featsT[((size_t)(h * 16 + d)) * NNODES + i] = f2bf(acc);
    // column 8 = 1.0 (Z accumulator), columns 9..15 = 0
    featsT[((size_t)(h * 16 + 8 + d)) * NNODES + i] =
        (d == 0) ? (unsigned short)0x3F80 : (unsigned short)0;

    __syncthreads();
    if (t < NH) {
        float s1 = 0.0f, s2 = 0.0f;
#pragma unroll
        for (int dd = 0; dd < ND; ++dd) {
            float f = frow[t * ND + dd];
            s1 = fmaf(f, att_self [t * ND + dd], s1);
            s2 = fmaf(f, att_neigh[t * ND + dd], s2);
        }
        a_sT[(size_t)t * NNODES + i] = s1 * LOG2E;   // pre-scale for exp2
        a_nT[(size_t)t * NNODES + i] = s2 * LOG2E;
    }
}

// ---------------------------------------------------------------------------
// Kernel 2: fused masked-softmax attention aggregation.
// Grid: 256 blocks (16-row tiles of i). Block: 256 threads = 8 waves,
// wave w handles head w. A[i-tile, j-chunk] + a_n[j-chunk] staged in LDS via
// async loads (double buffered), shared by all 8 heads. Per j-chunk: build
// bf16 M-tile in VALU, one v_wmma_f32_16x16x32_bf16 against featsT.
// ---------------------------------------------------------------------------
__global__ __launch_bounds__(256)
void gat_aggregate_kernel(const float* __restrict__ A,
                          const unsigned short* __restrict__ featsT,
                          const float* __restrict__ a_sT,
                          const float* __restrict__ a_nT,
                          const float* __restrict__ bias,
                          float* __restrict__ out) {
    const int tile = blockIdx.x;
    const int i0   = tile * 16;
    const int lane = threadIdx.x & 31;
    const int h    = threadIdx.x >> 5;   // wave id == head
    const int s    = lane >> 4;          // lane-half select
    const int r    = lane & 15;          // A-operand row / B-operand column

    __shared__ float Atile [2][16][32];  // double-buffered fp32 A tile
    __shared__ float Antile[2][NH][32];  // double-buffered a_n chunk per head

    // cooperative staging: 256 threads x 2 floats = 16x32 A tile
    const int t    = threadIdx.x;
    const int srow = t >> 4;
    const int scol = (t << 1) & 31;

    // per-thread LDS byte offsets for the async destinations
    const unsigned ldsA0 = (unsigned)(uintptr_t)(void*)&Atile [0][srow][scol];
    const unsigned ldsA1 = (unsigned)(uintptr_t)(void*)&Atile [1][srow][scol];
    const unsigned ldsN0 = (unsigned)(uintptr_t)(void*)&Antile[0][h][lane];
    const unsigned ldsN1 = (unsigned)(uintptr_t)(void*)&Antile[1][h][lane];

    const float* gA = A    + (size_t)(i0 + srow) * NNODES + scol;  // + jb
    const float* gN = a_nT + (size_t)h * NNODES + lane;            // + jb

    const float asr = a_sT[(size_t)h * NNODES + i0 + r];  // log2e-scaled

    v8f c = {};   // fp32 accumulator: cols 0..7 = out, col 8 = Z

    // prologue: stage tile 0 asynchronously
    async_lds_b64(ldsA0, gA);
    async_lds_b32(ldsN0, gN);
    wait_async0();
    __syncthreads();

    const int NITER = NNODES / 32;       // 128
    for (int it = 0; it < NITER; ++it) {
        const int jb  = it << 5;
        const int cur = it & 1;

        // issue async staging of the next tile; completion checked at the
        // bottom of the loop so it overlaps the exp/WMMA work below
        if (it + 1 < NITER) {
            async_lds_b64(cur ? ldsA0 : ldsA1, gA + (jb + 32));
            async_lds_b32(cur ? ldsN0 : ldsN1, gN + (jb + 32));
        }
        if (it + 2 < NITER) {
            // speculative stream prefetch of the tile after next into L2
            __builtin_prefetch(gA + (jb + 64), 0, 1);
        }

        // ---- build A-operand (M-tile) in contiguous k order (same as B) ----
        // lanes 0-15 : element e -> K = e
        // lanes 16-31: element e -> K = e+16
        const float* arow = &Atile [cur][r][0];
        const float* anp  = &Antile[cur][h][0];

        float4 a0 = *(const float4*)(arow + 16 * s);
        float4 a1 = *(const float4*)(arow + 16 * s + 4);
        float4 a2 = *(const float4*)(arow + 16 * s + 8);
        float4 a3 = *(const float4*)(arow + 16 * s + 12);
        float4 n0 = *(const float4*)(anp + 16 * s);
        float4 n1 = *(const float4*)(anp + 16 * s + 4);
        float4 n2 = *(const float4*)(anp + 16 * s + 8);
        float4 n3 = *(const float4*)(anp + 16 * s + 12);

        v16bf av;
        av[0]  = melem(a0.x, asr + n0.x);
        av[1]  = melem(a0.y, asr + n0.y);
        av[2]  = melem(a0.z, asr + n0.z);
        av[3]  = melem(a0.w, asr + n0.w);
        av[4]  = melem(a1.x, asr + n1.x);
        av[5]  = melem(a1.y, asr + n1.y);
        av[6]  = melem(a1.z, asr + n1.z);
        av[7]  = melem(a1.w, asr + n1.w);
        av[8]  = melem(a2.x, asr + n2.x);
        av[9]  = melem(a2.y, asr + n2.y);
        av[10] = melem(a2.z, asr + n2.z);
        av[11] = melem(a2.w, asr + n2.w);
        av[12] = melem(a3.x, asr + n3.x);
        av[13] = melem(a3.y, asr + n3.y);
        av[14] = melem(a3.z, asr + n3.z);
        av[15] = melem(a3.w, asr + n3.w);

        // ---- B-operand: 16 consecutive bf16 along j from featsT ----
        // column = r, K = 16*s + e  -> contiguous 32-byte load
        const unsigned short* bp =
            featsT + ((size_t)(h * 16 + r)) * NNODES + jb + 16 * s;
        v16u  bu = *(const v16u*)bp;
        v16bf bv = __builtin_bit_cast(v16bf, bu);

        // D = A x B + C  (fp32 accum)
        c = __builtin_amdgcn_wmma_f32_16x16x32_bf16(
                false, av, false, bv, (short)0, c, false, false);

        // make this wave's async stores to the "next" buffer visible, then
        // block-wide barrier before anyone reads them
        wait_async0();
        __syncthreads();
    }

    // ---- epilogue: divide by Z (output column 8), bias, relu, store ----
    // C layout: element v, lane l -> row = v + 8*(l>>4), col = l&15
    float res[8];
#pragma unroll
    for (int v = 0; v < 8; ++v) {
        float zv = __shfl(c[v], (lane & 16) | 8, 32);   // Z for this row set
        res[v] = c[v] / zv;
    }
    if (r < ND) {
        const float b = bias[h * ND + r];
#pragma unroll
        for (int v = 0; v < 8; ++v) {
            int   i = i0 + v + 8 * s;
            float o = res[v] + b;
            o = o > 0.0f ? o : 0.0f;
            out[(size_t)i * (NH * ND) + h * ND + r] = o;
        }
    }
}

// ---------------------------------------------------------------------------
// Launch
// ---------------------------------------------------------------------------
extern "C" void kernel_launch(void* const* d_in, const int* in_sizes, int n_in,
                              void* d_out, int out_size, void* d_ws, size_t ws_size,
                              hipStream_t stream) {
    const float* X         = (const float*)d_in[0];  // [4096,128]
    const float* A         = (const float*)d_in[1];  // [4096,4096]
    const float* W         = (const float*)d_in[2];  // [128,64]
    const float* att_self  = (const float*)d_in[3];  // [1,8,8]
    const float* att_neigh = (const float*)d_in[4];  // [1,8,8]
    const float* bias      = (const float*)d_in[5];  // [1,8,8]
    float* out = (float*)d_out;                      // [4096,64]

    // workspace layout
    unsigned short* featsT = (unsigned short*)d_ws;                 // 8*16*4096 bf16 = 1 MB
    float* a_sT = (float*)((char*)d_ws + (size_t)NH * 16 * NNODES * 2);
    float* a_nT = a_sT + (size_t)NH * NNODES;

    gat_feats_kernel<<<NNODES, 64, 0, stream>>>(X, W, att_self, att_neigh,
                                                featsT, a_sT, a_nT);
    gat_aggregate_kernel<<<NNODES / 16, 256, 0, stream>>>(A, featsT, a_sT, a_nT,
                                                          bias, out);
}
